// TaskGNN_23270132810526
// MI455X (gfx1250) — compile-verified, matrix operand order checked by reading, not run on hardware
//
#include <hip/hip_runtime.h>
#include <hip/hip_bf16.h>

// ---------------------------------------------------------------------------
// GAT (2 layers, HEADS=2, HID=64) + global mean pool + 3 softmax heads.
//  - Dense GEMMs: v_wmma_f32_16x16x32_f16, W staged to LDS via
//    global_load_async_to_lds_b128 (ASYNCcnt) and pre-swizzled to B-fragment
//    order (one contiguous 32B ds read per lane per fragment).
//  - Edge phases (bandwidth bound): float4 gathers + global_atomic_add_f32.
// ---------------------------------------------------------------------------

typedef __attribute__((ext_vector_type(16))) _Float16 v16h;
typedef __attribute__((ext_vector_type(8)))  float    v8f;

#define HEADS 2
#define HID   64
#define FDIM  (HEADS * HID)   // 128
#define NEG_SLOPE 0.2f

// ----- order-preserving float <-> uint key (for atomic segment-max) --------
__device__ __forceinline__ unsigned f2key(float f) {
    unsigned u = __float_as_uint(f);
    return (u & 0x80000000u) ? ~u : (u | 0x80000000u);
}
__device__ __forceinline__ float key2f(unsigned k) {
    unsigned u = (k & 0x80000000u) ? (k & 0x7FFFFFFFu) : ~k;
    return __uint_as_float(u);
}
__device__ __forceinline__ float lrelu(float v) {
    return v > 0.f ? v : NEG_SLOPE * v;
}

// ---------------------------------------------------------------------------
// GEMM: C[M,128] = A[M,K] * B[K,128], f32 data, f16 WMMA multiply, f32 accum.
// Block = 256 threads (8 waves); block computes a 128-row x 128-col strip.
// Wave w owns rows [blk*128 + w*16, +16).  K in {64,128}, multiple of 32.
//
// B is staged to LDS once per block, 32 K-rows at a time:
//   raw f32 chunk arrives via global_load_async_to_lds_b128 (ASYNCcnt),
//   then converted to f16 in WMMA B-fragment order:
//     swzW[ kb*4096 + ((t*32 + lane)*16 + i) ]
//   where t = 16-col tile, i = v16h element, and the element's K offset is
//     koff = 8*(lane>=16) + (i<8 ? i : i+8), column n = t*16 + (lane&15).
//
// CDNA5 16-bit A fragment (16x32 MxK), wave32:
//   lanes 0-15 : M = lane,    K = {0..7, 16..23}
//   lanes 16-31: M = lane-16, K = {8..15, 24..31}
//   -> af[0..7] / af[8..15] are two contiguous 8-float runs of the A row.
// f32 C/D: VGPR r -> row = r + 8*(lane>=16), col = lane&15.
// ---------------------------------------------------------------------------
__global__ __launch_bounds__(256) void gemm_wmma_f16(
    const float* __restrict__ A, const float* __restrict__ B,
    float* __restrict__ C, int M, int K)
{
    __shared__ __align__(16) float    rawW[32 * FDIM];          // 16 KB chunk
    __shared__ __align__(32) _Float16 swzW[4 * 8 * 32 * 16];    // 32 KB frags

    const int tid   = threadIdx.x;
    const int lane  = tid & 31;
    const int wave  = tid >> 5;
    const int kHalf = (lane >> 4) * 8;          // 0 or 8

    // ---- stage B into LDS (async copy + swizzle-convert), chunk per kb ----
    const int nkb = K >> 5;                     // K/32 chunks
    for (int kb = 0; kb < nkb; ++kb) {
        const unsigned chunkBytes = 32u * FDIM * 4u;            // 16 KB
        const unsigned ldsBase = (unsigned)(size_t)&rawW[0];
        const char*    gsrc    = (const char*)(B + (size_t)kb * 32 * FDIM);
        for (unsigned o = (unsigned)tid * 16u; o < chunkBytes; o += 256u * 16u) {
            unsigned           laddr = ldsBase + o;
            unsigned long long gaddr = (unsigned long long)(gsrc + o);
            asm volatile("global_load_async_to_lds_b128 %0, %1, off"
                         :: "v"(laddr), "v"(gaddr) : "memory");
        }
        asm volatile("s_wait_asynccnt 0" ::: "memory");
        __syncthreads();
        // swizzle-convert this 32-K chunk into fragment order
        for (int j = tid; j < 8 * 32 * 16; j += 256) {
            const int i = j & 15, l = (j >> 4) & 31, t = j >> 9;
            const int koff = ((l >> 4) * 8) + ((i < 8) ? i : (i + 8));
            const int n    = t * 16 + (l & 15);
            swzW[kb * 4096 + j] = (_Float16)rawW[koff * FDIM + n];
        }
        __syncthreads();
    }

    // ---- compute: each wave does 16 rows x 128 cols -----------------------
    int mrow = blockIdx.x * 128 + wave * 16 + (lane & 15);
    if (mrow >= M) mrow = M - 1;                // clamp: EXEC stays all-1s
    const float* arow = A + (size_t)mrow * K;

    v8f acc[8];
#pragma unroll
    for (int t = 0; t < 8; ++t) acc[t] = (v8f){0.f,0.f,0.f,0.f,0.f,0.f,0.f,0.f};

    for (int kb = 0; kb < nkb; ++kb) {
        const int k0 = kb * 32;
        if (k0 + 32 < K)
            __builtin_prefetch(arow + k0 + 32, 0, 0);   // global_prefetch_b8
        // A fragment: two contiguous 8-float runs -> 4x global_load_b128
        const float4 r0 = *(const float4*)(arow + k0 + kHalf);
        const float4 r1 = *(const float4*)(arow + k0 + kHalf + 4);
        const float4 r2 = *(const float4*)(arow + k0 + kHalf + 16);
        const float4 r3 = *(const float4*)(arow + k0 + kHalf + 20);
        v16h af;
        af[0]=(_Float16)r0.x; af[1]=(_Float16)r0.y; af[2]=(_Float16)r0.z; af[3]=(_Float16)r0.w;
        af[4]=(_Float16)r1.x; af[5]=(_Float16)r1.y; af[6]=(_Float16)r1.z; af[7]=(_Float16)r1.w;
        af[8]=(_Float16)r2.x; af[9]=(_Float16)r2.y; af[10]=(_Float16)r2.z; af[11]=(_Float16)r2.w;
        af[12]=(_Float16)r3.x; af[13]=(_Float16)r3.y; af[14]=(_Float16)r3.z; af[15]=(_Float16)r3.w;
#pragma unroll
        for (int t = 0; t < 8; ++t) {
            const v16h bf = *(const v16h*)&swzW[kb * 4096 + (t * 32 + lane) * 16];
            acc[t] = __builtin_amdgcn_wmma_f32_16x16x32_f16(
                false, af, false, bf, (short)0, acc[t], false, false);
        }
    }

    // ---- store ------------------------------------------------------------
    const int rbase = (lane >> 4) * 8;
#pragma unroll
    for (int t = 0; t < 8; ++t) {
        const int col = t * 16 + (lane & 15);
#pragma unroll
        for (int r = 0; r < 8; ++r) {
            const int row = blockIdx.x * 128 + wave * 16 + rbase + r;
            if (row < M)
                C[(size_t)row * FDIM + col] = acc[t][r];
        }
    }
}

// ---------------------------------------------------------------------------
// a_src[n,h] = sum_f h[n,h,f]*att_src[h,f] ; a_dst likewise.  1 thread/(n,h).
// ---------------------------------------------------------------------------
__global__ void att_dot_kernel(const float* __restrict__ h,
                               const float* __restrict__ att_src,
                               const float* __restrict__ att_dst,
                               float* __restrict__ a_src,
                               float* __restrict__ a_dst, int N)
{
    const int t = blockIdx.x * blockDim.x + threadIdx.x;
    if (t >= N * HEADS) return;
    const int n = t >> 1, hd = t & 1;
    const float* hp = h + (size_t)n * FDIM + hd * HID;
    const float* as = att_src + hd * HID;
    const float* ad = att_dst + hd * HID;
    float s = 0.f, d = 0.f;
#pragma unroll 8
    for (int f = 0; f < HID; ++f) { const float v = hp[f]; s += v * as[f]; d += v * ad[f]; }
    a_src[t] = s;
    a_dst[t] = d;
}

__device__ __forceinline__ void edge_sd(const long long* ei, int E, int e,
                                        long long& s, long long& d)
{
    if (e < E) { s = ei[e]; d = ei[(size_t)E + e]; }
    else       { s = d = (long long)(e - E); }          // self loop
}

// ---------------------------------------------------------------------------
// Pass 1: segment max of leaky_relu(a_src[src]+a_dst[dst]) keyed to dst.
// ---------------------------------------------------------------------------
__global__ void edge_max_kernel(const long long* __restrict__ ei, int E, int Etot,
                                const float* __restrict__ a_src,
                                const float* __restrict__ a_dst,
                                unsigned* __restrict__ mkey)
{
    const int e = blockIdx.x * blockDim.x + threadIdx.x;
    if (e >= Etot) return;
    long long s, d; edge_sd(ei, E, e, s, d);
#pragma unroll
    for (int hd = 0; hd < HEADS; ++hd) {
        const float v = lrelu(a_src[s * HEADS + hd] + a_dst[d * HEADS + hd]);
        atomicMax(&mkey[d * HEADS + hd], f2key(v));
    }
}

// ---------------------------------------------------------------------------
// Pass 2: expval[e,h] = exp(e - max[dst]); denom[dst,h] += expval.
// ---------------------------------------------------------------------------
__global__ void edge_exp_kernel(const long long* __restrict__ ei, int E, int Etot,
                                const float* __restrict__ a_src,
                                const float* __restrict__ a_dst,
                                const unsigned* __restrict__ mkey,
                                float* __restrict__ expval,
                                float* __restrict__ denom)
{
    const int e = blockIdx.x * blockDim.x + threadIdx.x;
    if (e >= Etot) return;
    long long s, d; edge_sd(ei, E, e, s, d);
#pragma unroll
    for (int hd = 0; hd < HEADS; ++hd) {
        const float v  = lrelu(a_src[s * HEADS + hd] + a_dst[d * HEADS + hd]);
        const float m  = key2f(mkey[d * HEADS + hd]);
        const float ex = __expf(v - m);
        expval[(size_t)e * HEADS + hd] = ex;
        atomicAdd(&denom[d * HEADS + hd], ex);
    }
}

// ---------------------------------------------------------------------------
// Pass 3: acc[dst,:,:] += h[src,:,:] * alpha.  One wave per edge; lane covers
// 4 contiguous features (float4 gather, 4 atomic f32 adds).
// ---------------------------------------------------------------------------
__global__ void edge_msg_kernel(const long long* __restrict__ ei, int E, int Etot,
                                const float* __restrict__ h,
                                const float* __restrict__ expval,
                                const float* __restrict__ denom,
                                float* __restrict__ acc)
{
    const long long t = (long long)blockIdx.x * blockDim.x + threadIdx.x;
    const int e    = (int)(t >> 5);
    const int lane = (int)(t & 31);
    if (e >= Etot) return;
    long long s, d; edge_sd(ei, E, e, s, d);

    const int idx = lane * 4;           // 0..124, never straddles a head
    const int hd  = idx >> 6;
    const float alpha = expval[(size_t)e * HEADS + hd] / denom[d * HEADS + hd];

    const float4 hv = *(const float4*)(h + (size_t)s * FDIM + idx);
    float* ap = acc + (size_t)d * FDIM + idx;
    atomicAdd(ap + 0, hv.x * alpha);
    atomicAdd(ap + 1, hv.y * alpha);
    atomicAdd(ap + 2, hv.z * alpha);
    atomicAdd(ap + 3, hv.w * alpha);
}

// ---------------------------------------------------------------------------
// out[n,f] = relu(mean_heads(acc) + bias).  1 thread / (n,f).
// ---------------------------------------------------------------------------
__global__ void node_out_kernel(const float* __restrict__ acc,
                                const float* __restrict__ bias,
                                float* __restrict__ out, int N)
{
    const int t = blockIdx.x * blockDim.x + threadIdx.x;
    if (t >= N * HID) return;
    const int n = t >> 6, f = t & 63;
    float v = 0.5f * (acc[(size_t)n * FDIM + f] + acc[(size_t)n * FDIM + HID + f]) + bias[f];
    out[t] = v > 0.f ? v : 0.f;
}

// ---------------------------------------------------------------------------
// Global mean pool: scatter sums + counts.
// ---------------------------------------------------------------------------
__global__ void pool_kernel(const float* __restrict__ x,
                            const long long* __restrict__ batch,
                            float* __restrict__ pooled,
                            float* __restrict__ counts, int N)
{
    const int t = blockIdx.x * blockDim.x + threadIdx.x;
    if (t >= N * HID) return;
    const int n = t >> 6, f = t & 63;
    const long long g = batch[n];
    atomicAdd(&pooled[(size_t)g * HID + f], x[t]);
    if (f == 0) atomicAdd(&counts[g], 1.0f);
}

// ---------------------------------------------------------------------------
// Per-graph heads: logits + softmax for object(32)/goal(16)/action(8).
// One 64-thread block per graph.
// ---------------------------------------------------------------------------
__global__ __launch_bounds__(64) void head_kernel(
    const float* __restrict__ pooled, const float* __restrict__ counts,
    const float* __restrict__ Wo, const float* __restrict__ bo,
    const float* __restrict__ Wg, const float* __restrict__ bg,
    const float* __restrict__ Wa, const float* __restrict__ ba,
    float* __restrict__ out, int G)
{
    __shared__ float p[HID];
    __shared__ float lo[32], lg[16], la[8];
    const int g = blockIdx.x, tid = threadIdx.x;

    float c = counts[g];
    c = c > 1.f ? c : 1.f;
    p[tid] = pooled[(size_t)g * HID + tid] / c;
    __syncthreads();

    if (tid < 32) { float s = bo[tid]; for (int k = 0; k < HID; ++k) s += p[k] * Wo[k * 32 + tid]; lo[tid] = s; }
    if (tid < 16) { float s = bg[tid]; for (int k = 0; k < HID; ++k) s += p[k] * Wg[k * 16 + tid]; lg[tid] = s; }
    if (tid <  8) { float s = ba[tid]; for (int k = 0; k < HID; ++k) s += p[k] * Wa[k *  8 + tid]; la[tid] = s; }
    __syncthreads();

    if (tid == 0) {
        float* dsts[3] = { out + (size_t)g * 32,
                           out + (size_t)G * 32 + (size_t)g * 16,
                           out + (size_t)G * 48 + (size_t)g * 8 };
        float* srcs[3] = { lo, lg, la };
        const int lens[3] = { 32, 16, 8 };
        for (int h = 0; h < 3; ++h) {
            float m = srcs[h][0];
            for (int j = 1; j < lens[h]; ++j) m = srcs[h][j] > m ? srcs[h][j] : m;
            float sum = 0.f;
            for (int j = 0; j < lens[h]; ++j) { const float e = __expf(srcs[h][j] - m); srcs[h][j] = e; sum += e; }
            for (int j = 0; j < lens[h]; ++j) dsts[h][j] = srcs[h][j] / sum;
        }
    }
}

// ---------------------------------------------------------------------------
extern "C" void kernel_launch(void* const* d_in, const int* in_sizes, int n_in,
                              void* d_out, int out_size, void* d_ws, size_t ws_size,
                              hipStream_t stream)
{
    const float*     x     = (const float*)d_in[0];
    const long long* ei    = (const long long*)d_in[1];   // int64 [2,E]
    const long long* batch = (const long long*)d_in[2];   // int64 [N]
    const float* W1   = (const float*)d_in[3];
    const float* as1  = (const float*)d_in[4];
    const float* ad1  = (const float*)d_in[5];
    const float* b1   = (const float*)d_in[6];
    const float* W2   = (const float*)d_in[7];
    const float* as2  = (const float*)d_in[8];
    const float* ad2  = (const float*)d_in[9];
    const float* b2   = (const float*)d_in[10];
    const float* Wo   = (const float*)d_in[11];
    const float* bo   = (const float*)d_in[12];
    const float* Wg   = (const float*)d_in[13];
    const float* bg   = (const float*)d_in[14];
    const float* Wa   = (const float*)d_in[15];
    const float* ba   = (const float*)d_in[16];
    float* outp = (float*)d_out;

    const int N    = in_sizes[2];
    const int E    = in_sizes[1] / 2;
    const int Etot = E + N;
    const int G    = out_size / 56;   // 32 + 16 + 8 per graph

    // ----- workspace layout -----
    char* ws = (char*)d_ws;
    size_t off = 0;
    auto take = [&](size_t bytes) -> char* {
        char* p = ws + off;
        off = (off + bytes + 255) & ~(size_t)255;
        return p;
    };
    float*    bufH   = (float*)   take((size_t)N * FDIM * 4);   // h (both layers)
    float*    bufX   = (float*)   take((size_t)N * HID  * 4);   // layer out / next in
    float*    accM   = (float*)   take((size_t)N * FDIM * 4);   // message accum
    float*    aSrc   = (float*)   take((size_t)N * HEADS * 4);
    float*    aDst   = (float*)   take((size_t)N * HEADS * 4);
    unsigned* mkey   = (unsigned*)take((size_t)N * HEADS * 4);
    float*    denom  = (float*)   take((size_t)N * HEADS * 4);
    float*    expv   = (float*)   take((size_t)Etot * HEADS * 4);
    float*    pooled = (float*)   take((size_t)G * HID * 4);
    float*    counts = (float*)   take((size_t)G * 4);
    (void)ws_size;

    const int TB = 256;
    const dim3 gEdge((Etot + TB - 1) / TB);
    const dim3 gMsg((unsigned)(((size_t)Etot * 32 + TB - 1) / TB));
    const dim3 gNF((N * HID + TB - 1) / TB);
    const dim3 gNH((N * HEADS + TB - 1) / TB);
    const dim3 gGemm((N + 127) / 128);

    // =================== Layer 1 ===================
    gemm_wmma_f16<<<gGemm, 256, 0, stream>>>(x, W1, bufH, N, 128);
    att_dot_kernel<<<gNH, TB, 0, stream>>>(bufH, as1, ad1, aSrc, aDst, N);

    hipMemsetAsync(mkey,  0, (size_t)N * HEADS * 4, stream);
    edge_max_kernel<<<gEdge, TB, 0, stream>>>(ei, E, Etot, aSrc, aDst, mkey);

    hipMemsetAsync(denom, 0, (size_t)N * HEADS * 4, stream);
    edge_exp_kernel<<<gEdge, TB, 0, stream>>>(ei, E, Etot, aSrc, aDst, mkey, expv, denom);

    hipMemsetAsync(accM, 0, (size_t)N * FDIM * 4, stream);
    edge_msg_kernel<<<gMsg, TB, 0, stream>>>(ei, E, Etot, bufH, expv, denom, accM);

    node_out_kernel<<<gNF, TB, 0, stream>>>(accM, b1, bufX, N);

    // =================== Layer 2 ===================
    gemm_wmma_f16<<<gGemm, 256, 0, stream>>>(bufX, W2, bufH, N, HID);
    att_dot_kernel<<<gNH, TB, 0, stream>>>(bufH, as2, ad2, aSrc, aDst, N);

    hipMemsetAsync(mkey,  0, (size_t)N * HEADS * 4, stream);
    edge_max_kernel<<<gEdge, TB, 0, stream>>>(ei, E, Etot, aSrc, aDst, mkey);

    hipMemsetAsync(denom, 0, (size_t)N * HEADS * 4, stream);
    edge_exp_kernel<<<gEdge, TB, 0, stream>>>(ei, E, Etot, aSrc, aDst, mkey, expv, denom);

    hipMemsetAsync(accM, 0, (size_t)N * FDIM * 4, stream);
    edge_msg_kernel<<<gMsg, TB, 0, stream>>>(ei, E, Etot, bufH, expv, denom, accM);

    node_out_kernel<<<gNF, TB, 0, stream>>>(accM, b2, bufX, N);

    // =================== Pool + heads ===================
    hipMemsetAsync(pooled, 0, (size_t)G * HID * 4, stream);
    hipMemsetAsync(counts, 0, (size_t)G * 4, stream);
    pool_kernel<<<gNF, TB, 0, stream>>>(bufX, batch, pooled, counts, N);

    head_kernel<<<G, 64, 0, stream>>>(pooled, counts, Wo, bo, Wg, bg, Wa, ba, outp, G);
}
